// Attention_24781961298420
// MI455X (gfx1250) — matrix operand order repaired and checked by reading, not compiled, not run on hardware
//
#include <hip/hip_runtime.h>

typedef __attribute__((ext_vector_type(16))) __bf16 v16bf;
typedef __attribute__((ext_vector_type(8)))  __bf16 v8bf;
typedef __attribute__((ext_vector_type(8)))  float  v8f;

union V16 { v16bf v; v8bf h[2]; };

// Problem constants (match reference)
constexpr int S  = 2048;
constexpr int D  = 2048;
constexpr int H  = 32;
constexpr int KV = 4;
constexpr int HD = 64;
constexpr int NREP = H / KV; // 8

// ---------------------------------------------------------------------------
// CDNA5 async global->LDS copy (ASYNCcnt-tracked) + wait
// ---------------------------------------------------------------------------
__device__ inline void async_ld_b128(unsigned int lds_off, const void* gptr) {
  asm volatile("global_load_async_to_lds_b128 %0, %1, off"
               :: "v"(lds_off), "v"(gptr) : "memory");
}
__device__ inline void wait_async0() {
#if __has_builtin(__builtin_amdgcn_s_wait_asynccnt)
  __builtin_amdgcn_s_wait_asynccnt(0);
#else
  asm volatile("s_wait_asynccnt 0" ::: "memory");
#endif
}
__device__ inline unsigned int lds_off32(const void* p) {
  return (unsigned int)(size_t)p; // low 32 bits = LDS byte address (ISA 10.2)
}

// ---------------------------------------------------------------------------
// fp32 -> bf16 elementwise conversion
// ---------------------------------------------------------------------------
__global__ __launch_bounds__(256) void cvt_f32_bf16(const float* __restrict__ in,
                                                    __bf16* __restrict__ out, int n) {
  int i = blockIdx.x * 256 + threadIdx.x;
  if (i < n) out[i] = (__bf16)in[i];
}

// ---------------------------------------------------------------------------
// RoPE: in (S, nh, HD) f32 -> out (S, nh, HD) bf16, interleaved pair rotation
// ---------------------------------------------------------------------------
__global__ __launch_bounds__(256) void rope_cvt(const float* __restrict__ in,
                                                __bf16* __restrict__ out,
                                                const float* __restrict__ cs,
                                                const float* __restrict__ sn,
                                                int nheads, int total_pairs) {
  int i = blockIdx.x * 256 + threadIdx.x;
  if (i >= total_pairs) return;
  int p = i & 31;            // HD/2 = 32 pairs
  int t = i >> 5;
  int hh = t % nheads;
  int s  = t / nheads;
  float c  = cs[s * 32 + p];
  float si = sn[s * 32 + p];
  size_t base = ((size_t)s * nheads + hh) * HD + 2 * p;
  float t0 = in[base], t1 = in[base + 1];
  out[base]     = (__bf16)(t0 * c - t1 * si);
  out[base + 1] = (__bf16)(t0 * si + t1 * c);
}

// ---------------------------------------------------------------------------
// Tiled bf16 GEMM: C[M,N] = A[M,K] * B[K,N], A/B bf16 row-major, C f32.
// Block tile 128x128, 8 waves as 4(M) x 2(N); each wave computes 2x4 WMMA
// tiles (32x64).  Double-buffered LDS: the next K-slab's A tile is fetched
// with global_load_async_to_lds_b128 and B is transposed into LDS while the
// current slab's 8 WMMAs execute; one s_wait_asynccnt + barrier per slab.
// Requires M % 128 == 0, N % 128 == 0, K % 32 == 0 (true for all uses).
// ---------------------------------------------------------------------------
__global__ __launch_bounds__(256) void gemm_bf16(const __bf16* __restrict__ A,
                                                 const __bf16* __restrict__ B,
                                                 float* __restrict__ C,
                                                 int M, int N, int K) {
  __shared__ __bf16 As[2][128][40];   // [buf][m][k], 80B row stride
  __shared__ __bf16 Bt[2][128][40];   // [buf][n][k], transposed B tile
  const int tid  = threadIdx.x;
  const int lane = tid & 31;
  const int w    = tid >> 5;
  const int wm   = w & 3;   // row group (32 rows each)
  const int wn   = w >> 2;  // col group (64 cols each)
  const int rowBlk = blockIdx.y * 128;
  const int colBlk = blockIdx.x * 128;

  v8f acc[2][4] = {};

  const int arow  = lane & 15;
  const int kb    = (lane >> 4) * 8;    // A-fragment K base
  const int bcol  = lane & 15;
  const int kbase = (lane >> 4) * 8;    // B-fragment K base

  // B-transpose staging map: 4 cols x 2 ks per thread, two k sub-passes
  const int n4 = (tid & 31) * 4;        // 0..124
  const int kp = (tid >> 5) * 2;        // 0..14

  auto stage = [&](int buf, int k0) {
    // A tile 128x32 via async global->LDS b128 (2 chunks/thread)
    #pragma unroll
    for (int i = 0; i < 2; ++i) {
      int chunk = tid + i * 256;        // 512 x 16B chunks
      int r  = chunk >> 2;
      int kk = (chunk & 3) * 8;
      async_ld_b128(lds_off32(&As[buf][r][kk]),
                    A + (size_t)(rowBlk + r) * K + k0 + kk);
    }
    // B tile 32x128 transposed into Bt[n][k] (pair-packed b32 stores)
    #pragma unroll
    for (int koff = 0; koff < 32; koff += 16) {
      int k2 = kp + koff;
      const __bf16* b0 = B + (size_t)(k0 + k2) * N + colBlk + n4;
      const __bf16* b1 = b0 + N;
      uint2 lo = *(const uint2*)b0;
      uint2 hi = *(const uint2*)b1;
      const unsigned short* ls = (const unsigned short*)&lo;
      const unsigned short* hs = (const unsigned short*)&hi;
      #pragma unroll
      for (int j = 0; j < 4; ++j) {
        unsigned int pv = (unsigned int)ls[j] | ((unsigned int)hs[j] << 16);
        *(unsigned int*)&Bt[buf][n4 + j][k2] = pv;
      }
    }
  };

  // Prologue: fill buffer 0
  stage(0, 0);
  wait_async0();
  __syncthreads();

  for (int k0 = 0; k0 < K; k0 += 32) {
    const int cur = (k0 >> 5) & 1;
    // Pipeline: kick off next slab while this one computes
    if (k0 + 32 < K) stage(cur ^ 1, k0 + 32);
    if (k0 + 64 < K) { // hint slab after next toward L2 (global_prefetch_b8)
      __builtin_prefetch(A + (size_t)(rowBlk + (tid >> 1)) * K + k0 + 64, 0, 1);
    }

    // ---- Fragments: all contiguous 16B-aligned ds_load_b128 reads
    v16bf af[2];
    #pragma unroll
    for (int mt = 0; mt < 2; ++mt) {
      int r = wm * 32 + mt * 16 + arow;
      V16 a;
      a.h[0] = *(const v8bf*)&As[cur][r][kb];
      a.h[1] = *(const v8bf*)&As[cur][r][16 + kb];
      af[mt] = a.v;
    }
    #pragma unroll
    for (int nt = 0; nt < 4; ++nt) {
      int c = wn * 64 + nt * 16 + bcol;
      V16 b;
      b.h[0] = *(const v8bf*)&Bt[cur][c][kbase];
      b.h[1] = *(const v8bf*)&Bt[cur][c][kbase + 16];
      #pragma unroll
      for (int mt = 0; mt < 2; ++mt)
        acc[mt][nt] = __builtin_amdgcn_wmma_f32_16x16x32_bf16(
            false, af[mt], false, b.v, (short)0, acc[mt][nt], false, false);
    }

    wait_async0();      // own async writes for next slab have landed
    __syncthreads();    // everyone done reading cur / writing next
  }

  // Store C (C/D layout: lanes 0-15 -> M=v, lanes 16-31 -> M=v+8; col = lane&15)
  const int ccol = lane & 15;
  const int mhi  = (lane >> 4) * 8;
  #pragma unroll
  for (int mt = 0; mt < 2; ++mt)
    #pragma unroll
    for (int nt = 0; nt < 4; ++nt)
      #pragma unroll
      for (int v = 0; v < 8; ++v) {
        int r = rowBlk + wm * 32 + mt * 16 + mhi + v;
        int c = colBlk + wn * 64 + nt * 16 + ccol;
        C[(size_t)r * N + c] = acc[mt][nt][v];
      }
}

// ---------------------------------------------------------------------------
// Cross-lane reductions within 16-lane halves (rows of a WMMA C tile)
// ---------------------------------------------------------------------------
__device__ inline float redmax16(float v) {
  #pragma unroll
  for (int m = 1; m < 16; m <<= 1) v = fmaxf(v, __shfl_xor(v, m, 32));
  return v;
}
__device__ inline float redsum16(float v) {
  #pragma unroll
  for (int m = 1; m < 16; m <<= 1) v += __shfl_xor(v, m, 32);
  return v;
}

// ---------------------------------------------------------------------------
// Flash attention (causal, GQA).  Block = 1 head x 128 query rows (8 waves,
// 16 rows each).  Key block of 32 staged verbatim via async b128 (natural
// [key][d] layout is exactly the QK^T B-fragment layout); V staged transposed
// for the PV step.  Online softmax in registers.
// ---------------------------------------------------------------------------
__global__ __launch_bounds__(256) void flash_attn(const __bf16* __restrict__ Qg,
                                                  const __bf16* __restrict__ Kg,
                                                  const __bf16* __restrict__ Vg,
                                                  __bf16* __restrict__ ctx) {
  __shared__ __bf16 Ks[32][72];      // [key][d]  (natural)
  __shared__ __bf16 Vt[64][40];      // [d][key]  (transposed)
  __shared__ __bf16 Ps[8][16][40];   // per-wave P re-layout patch

  const int tid  = threadIdx.x;
  const int lane = tid & 31;
  const int w    = tid >> 5;
  const int h    = blockIdx.y;
  const int kvh  = h / NREP;
  const int qb0  = blockIdx.x * 128;
  const int qr0  = qb0 + w * 16;

  const int col = lane & 15;
  const int mhi = (lane >> 4) * 8;
  const int kbA = (lane >> 4) * 8;    // A-fragment K base
  const int kbB = (lane >> 4) * 8;    // B-fragment K base

  // Load Q fragments for this wave's 16 rows (two K-steps over HD=64)
  v16bf qf[2];
  {
    int q = qr0 + (lane & 15);
    const __bf16* qrow = Qg + (size_t)q * (H * HD) + h * HD;
    #pragma unroll
    for (int ks = 0; ks < 2; ++ks) {
      V16 a;
      a.h[0] = *(const v8bf*)&qrow[ks * 32 + kbA];
      a.h[1] = *(const v8bf*)&qrow[ks * 32 + 16 + kbA];
      qf[ks] = a.v;
    }
  }

  v8f   o[4] = {};
  float mold[8], lsum[8];
  #pragma unroll
  for (int v = 0; v < 8; ++v) { mold[v] = -1e30f; lsum[v] = 0.0f; }

  // staging maps
  const int skey = tid >> 3;          // K chunk: key row
  const int sd0  = (tid & 7) * 8;     // K chunk: d offset (16B chunk)
  const int d4   = (tid & 15) * 4;    // V transpose: 4 d values
  const int key2 = (tid >> 4) * 2;    // V transpose: 2 keys

  const int kend = qb0 + 128;
  for (int kb = 0; kb < kend; kb += 32) {
    __syncthreads();
    // ---- K block verbatim via async b128 (1 chunk/thread)
    async_ld_b128(lds_off32(&Ks[skey][sd0]),
                  Kg + (size_t)(kb + skey) * (KV * HD) + kvh * HD + sd0);
    // ---- V block transposed into Vt[d][key] (pair-packed b32 stores)
    {
      const __bf16* v0 = Vg + (size_t)(kb + key2) * (KV * HD) + kvh * HD + d4;
      const __bf16* v1 = v0 + KV * HD;
      uint2 lo = *(const uint2*)v0;
      uint2 hi = *(const uint2*)v1;
      const unsigned short* ls = (const unsigned short*)&lo;
      const unsigned short* hs = (const unsigned short*)&hi;
      #pragma unroll
      for (int j = 0; j < 4; ++j) {
        unsigned int pv = (unsigned int)ls[j] | ((unsigned int)hs[j] << 16);
        *(unsigned int*)&Vt[d4 + j][key2] = pv;
      }
    }
    wait_async0();
    __syncthreads();

    // ---- S = Q K^T  (two 16x16 tiles over 32 keys, 2 K-steps over HD)
    v8f sc[2] = {};
    #pragma unroll
    for (int nt = 0; nt < 2; ++nt) {
      int c = nt * 16 + col;           // key column
      #pragma unroll
      for (int ks = 0; ks < 2; ++ks) {
        V16 b;
        b.h[0] = *(const v8bf*)&Ks[c][ks * 32 + kbB];
        b.h[1] = *(const v8bf*)&Ks[c][ks * 32 + kbB + 16];
        sc[nt] = __builtin_amdgcn_wmma_f32_16x16x32_bf16(
            false, qf[ks], false, b.v, (short)0, sc[nt], false, false);
      }
    }

    // ---- scale, causal mask, online softmax
    float corr[8];
    #pragma unroll
    for (int v = 0; v < 8; ++v) {
      int q = qr0 + mhi + v;
      float s0 = sc[0][v] * 0.125f;
      float s1 = sc[1][v] * 0.125f;
      if (kb + col      > q) s0 = -1e9f;
      if (kb + 16 + col > q) s1 = -1e9f;
      float rmax = redmax16(fmaxf(s0, s1));
      float mnew = fmaxf(mold[v], rmax);
      float cr   = __expf(mold[v] - mnew);
      float e0   = __expf(s0 - mnew);
      float e1   = __expf(s1 - mnew);
      lsum[v] = lsum[v] * cr + redsum16(e0 + e1);
      mold[v] = mnew;
      corr[v] = cr;
      Ps[w][mhi + v][col]      = (__bf16)e0;
      Ps[w][mhi + v][16 + col] = (__bf16)e1;
    }
    #pragma unroll
    for (int nt = 0; nt < 4; ++nt)
      #pragma unroll
      for (int v = 0; v < 8; ++v) o[nt][v] *= corr[v];

    // ---- P re-layout (C-tile -> A-fragment, same-wave LDS, DS in-order)
    v16bf pf;
    {
      int r = lane & 15;
      V16 p;
      p.h[0] = *(const v8bf*)&Ps[w][r][kbA];
      p.h[1] = *(const v8bf*)&Ps[w][r][16 + kbA];
      pf = p.v;
    }

    // ---- O += P V  (four 16-wide d tiles)
    #pragma unroll
    for (int nt = 0; nt < 4; ++nt) {
      int c = nt * 16 + col;           // d column
      V16 b;
      b.h[0] = *(const v8bf*)&Vt[c][kbB];
      b.h[1] = *(const v8bf*)&Vt[c][kbB + 16];
      o[nt] = __builtin_amdgcn_wmma_f32_16x16x32_bf16(
          false, pf, false, b.v, (short)0, o[nt], false, false);
    }
  }

  // ---- normalize and write context (bf16, feeds the wo GEMM)
  #pragma unroll
  for (int nt = 0; nt < 4; ++nt)
    #pragma unroll
    for (int v = 0; v < 8; ++v) {
      int q = qr0 + mhi + v;
      float val = o[nt][v] / lsum[v];
      ctx[(size_t)q * (H * HD) + h * HD + nt * 16 + col] = (__bf16)val;
    }
}

// ---------------------------------------------------------------------------
// Host-side orchestration
// ---------------------------------------------------------------------------
extern "C" void kernel_launch(void* const* d_in, const int* in_sizes, int n_in,
                              void* d_out, int out_size, void* d_ws, size_t ws_size,
                              hipStream_t stream) {
  const float* x    = (const float*)d_in[0];
  const float* fcos = (const float*)d_in[1];
  const float* fsin = (const float*)d_in[2];
  // d_in[3] = mask (causal structure computed analytically in-kernel)
  const float* wq = (const float*)d_in[4];
  const float* wk = (const float*)d_in[5];
  const float* wv = (const float*)d_in[6];
  const float* wo = (const float*)d_in[7];
  float* out = (float*)d_out;

  // Workspace carve-out (256B aligned)
  size_t off = 0;
  auto carve = [&](size_t bytes) {
    void* p = (char*)d_ws + off;
    off += (bytes + 255) & ~(size_t)255;
    return p;
  };
  const size_t SD  = (size_t)S * D;        // 2048*2048
  const size_t SKV = (size_t)S * KV * HD;  // 2048*256
  __bf16* xb   = (__bf16*)carve(SD * 2);
  __bf16* wqb  = (__bf16*)carve(SD * 2);
  __bf16* wkb  = (__bf16*)carve((size_t)D * KV * HD * 2);
  __bf16* wvb  = (__bf16*)carve((size_t)D * KV * HD * 2);
  __bf16* wob  = (__bf16*)carve(SD * 2);
  float*  Qf   = (float*) carve(SD * 4);
  float*  Kf   = (float*) carve(SKV * 4);
  float*  Vf   = (float*) carve(SKV * 4);
  __bf16* Qb   = (__bf16*)carve(SD * 2);
  __bf16* Kb   = (__bf16*)carve(SKV * 2);
  __bf16* Vb   = (__bf16*)carve(SKV * 2);
  __bf16* ctxb = (__bf16*)carve(SD * 2);
  (void)ws_size; (void)n_in; (void)in_sizes; (void)out_size;

  // 1) Convert inputs/weights to bf16
  cvt_f32_bf16<<<(int)((SD + 255) / 256), 256, 0, stream>>>(x,  xb,  (int)SD);
  cvt_f32_bf16<<<(int)((SD + 255) / 256), 256, 0, stream>>>(wq, wqb, (int)SD);
  cvt_f32_bf16<<<(int)((D * KV * HD + 255) / 256), 256, 0, stream>>>(wk, wkb, (int)(D * KV * HD));
  cvt_f32_bf16<<<(int)((D * KV * HD + 255) / 256), 256, 0, stream>>>(wv, wvb, (int)(D * KV * HD));
  cvt_f32_bf16<<<(int)((SD + 255) / 256), 256, 0, stream>>>(wo, wob, (int)SD);

  // 2) QKV projections (bf16 WMMA, f32 out)
  gemm_bf16<<<dim3(D / 128, S / 128), 256, 0, stream>>>(xb, wqb, Qf, S, D, D);
  gemm_bf16<<<dim3((KV * HD) / 128, S / 128), 256, 0, stream>>>(xb, wkb, Kf, S, KV * HD, D);
  gemm_bf16<<<dim3((KV * HD) / 128, S / 128), 256, 0, stream>>>(xb, wvb, Vf, S, KV * HD, D);

  // 3) RoPE(Q), RoPE(K), cvt(V)
  {
    int qpairs = S * H * (HD / 2);
    rope_cvt<<<(qpairs + 255) / 256, 256, 0, stream>>>(Qf, Qb, fcos, fsin, H, qpairs);
    int kpairs = S * KV * (HD / 2);
    rope_cvt<<<(kpairs + 255) / 256, 256, 0, stream>>>(Kf, Kb, fcos, fsin, KV, kpairs);
    cvt_f32_bf16<<<(int)((SKV + 255) / 256), 256, 0, stream>>>(Vf, Vb, (int)SKV);
  }

  // 4) Causal GQA flash attention -> ctx (bf16)
  flash_attn<<<dim3(S / 128, H), 256, 0, stream>>>(Qb, Kb, Vb, ctxb);

  // 5) Output projection -> d_out (f32)
  gemm_bf16<<<dim3(D / 128, S / 128), 256, 0, stream>>>(ctxb, wob, out, S, D, D);
}
